// Pooling_block_35545149341730
// MI455X (gfx1250) — compile-verified, matrix operand order checked by reading, not run on hardware
//
#include <hip/hip_runtime.h>
#include <hip/hip_bf16.h>

typedef __attribute__((ext_vector_type(16))) __bf16 v16bf;
typedef __attribute__((ext_vector_type(8)))  __bf16 bf8;
typedef __attribute__((ext_vector_type(8)))  float  v8f;

constexpr int kB   = 128;
constexpr int kN   = 785;     // 1 cls token + 784 grid
constexpr int kC   = 384;
constexpr int kH   = 28;
constexpr int kHP  = 14;
constexpr int kTOK = 197;     // 1 + 14*14
constexpr int kM   = kB * kTOK;   // 25216 rows of the final GEMM
constexpr int kNO  = 2 * kC;      // 768 output cols

__device__ __forceinline__ float sigmoidf_(float v) {
    return 1.0f / (1.0f + __expf(-v));
}

// ---- Kernel 0: convert W_out_cls (768x384, row-major, K contiguous) to bf16 ----
__global__ void wcvt_kernel(const float* __restrict__ W, __bf16* __restrict__ Wb, int n) {
    int i = blockIdx.x * blockDim.x + threadIdx.x;
    if (i < n) Wb[i] = (__bf16)W[i];
}

// ---- Kernel 1: s[b,c] = sigmoid(mean(edge[b,:,c])) + sigmoid(mean(x[b,1:,c])) ----
__global__ void agg_kernel(const float* __restrict__ x, const float* __restrict__ edge,
                           float* __restrict__ s) {
    int b = blockIdx.x;
    int c = threadIdx.x;
    const float* ep = edge + (size_t)b * kN * kC + c;
    const float* xp = x    + (size_t)b * kN * kC + c;
    float se = 0.f, sn = 0.f;
    for (int n = 0; n < kN; ++n) se += ep[(size_t)n * kC];
    for (int n = 1; n < kN; ++n) sn += xp[(size_t)n * kC];
    se *= (1.0f / (float)kN);
    sn *= (1.0f / (float)(kN - 1));
    s[b * kC + c] = sigmoidf_(se) + sigmoidf_(sn);
}

// ---- Kernel 2: ci[b,c] = sum_k s[b,k] * W_lin[c,k] ----
__global__ void ci_kernel(const float* __restrict__ s, const float* __restrict__ Wlin,
                          float* __restrict__ ci) {
    __shared__ float sh[kC];
    int b = blockIdx.x, c = threadIdx.x;
    sh[c] = s[b * kC + c];
    __syncthreads();
    const float* w = Wlin + (size_t)c * kC;
    float acc = 0.f;
    for (int k = 0; k < kC; ++k) acc += sh[k] * w[k];
    ci[b * kC + c] = acc;
}

// ---- Kernel 3: ssig[b,n] = sigmoid(sum_c x[b,n+1,c] * ci[b,c]), one wave per (b,n) ----
__global__ void score_kernel(const float* __restrict__ x, const float* __restrict__ ci,
                             float* __restrict__ ssig) {
    int wid  = (blockIdx.x * blockDim.x + threadIdx.x) >> 5;
    int lane = threadIdx.x & 31;
    if (wid >= kB * (kH * kH)) return;
    int b = wid / (kH * kH), n = wid % (kH * kH);
    const float* xp = x  + (size_t)b * kN * kC + (size_t)(n + 1) * kC;
    const float* cp = ci + (size_t)b * kC;
    float acc = 0.f;
    for (int c = lane; c < kC; c += 32) acc += xp[c] * cp[c];
    #pragma unroll
    for (int off = 16; off > 0; off >>= 1) acc += __shfl_xor(acc, off, 32);
    if (lane == 0) ssig[b * (kH * kH) + n] = sigmoidf_(acc);
}

// ---- Kernel 4: assemble mid (bf16): cls row + 2x2 pooled rows ----
__global__ void pool_kernel(const float* __restrict__ x, const float* __restrict__ ssig,
                            const float* __restrict__ weights, __bf16* __restrict__ mid) {
    int bt = blockIdx.x;
    int b = bt / kTOK, t = bt % kTOK, c = threadIdx.x;
    __bf16* mp = mid + (size_t)bt * kC;
    const float* xb = x + (size_t)b * kN * kC;
    if (t == 0) { mp[c] = (__bf16)xb[c]; return; }
    int k = t - 1, i = k / kHP, j = k % kHP;
    const float* sp = ssig + (size_t)b * (kH * kH);
    float pooled = 0.f;
    #pragma unroll
    for (int q = 0; q < 2; ++q) {
        float wsum = 0.f;
        #pragma unroll
        for (int r = 0; r < 2; ++r) {
            float sv = sp[(2 * i + q) * kH + (2 * j + r)];
            wsum += weights[(c * 2 + q) * 2 + r] * (sv + 1.0f);
        }
        float n0 = xb[(size_t)(1 + (2 * i + 0) * kH + (2 * j + q)) * kC + c];
        float n1 = xb[(size_t)(1 + (2 * i + 1) * kH + (2 * j + q)) * kC + c];
        pooled += (n0 + n1) * wsum;
    }
    mp[c] = (__bf16)pooled;
}

// ---- Kernel 5: out[r,m] = sum_k mid[r,k] * Wb[m,k]  via v_wmma_f32_16x16x32_bf16 ----
// One wave -> 16x64 output tile (A fragment reused across 4 column tiles).
__global__ void __launch_bounds__(256) gemm_kernel(const __bf16* __restrict__ A,
                                                   const __bf16* __restrict__ Bw,
                                                   float* __restrict__ out) {
    int wid  = (blockIdx.x * blockDim.x + threadIdx.x) >> 5;
    int lane = threadIdx.x & 31;
    int half = lane >> 4;          // 0: lanes 0-15, 1: lanes 16-31
    int lidx = lane & 15;
    int row_tile = wid / (kNO / 64);
    int cg       = wid % (kNO / 64);
    int row0 = row_tile * 16;
    int col0 = cg * 64;

    const __bf16* ap  = A  + (size_t)(row0 + lidx) * kC + half * 8;
    const __bf16* bp0 = Bw + (size_t)(col0 +  0 + lidx) * kC + half * 8;
    const __bf16* bp1 = Bw + (size_t)(col0 + 16 + lidx) * kC + half * 8;
    const __bf16* bp2 = Bw + (size_t)(col0 + 32 + lidx) * kC + half * 8;
    const __bf16* bp3 = Bw + (size_t)(col0 + 48 + lidx) * kC + half * 8;

    v8f acc0 = {}, acc1 = {}, acc2 = {}, acc3 = {};

    for (int k = 0; k < kC; k += 32) {
        bf8 alo = *(const bf8*)(ap  + k);
        bf8 ahi = *(const bf8*)(ap  + k + 16);
        bf8 b0l = *(const bf8*)(bp0 + k);
        bf8 b0h = *(const bf8*)(bp0 + k + 16);
        bf8 b1l = *(const bf8*)(bp1 + k);
        bf8 b1h = *(const bf8*)(bp1 + k + 16);
        bf8 b2l = *(const bf8*)(bp2 + k);
        bf8 b2h = *(const bf8*)(bp2 + k + 16);
        bf8 b3l = *(const bf8*)(bp3 + k);
        bf8 b3h = *(const bf8*)(bp3 + k + 16);

        v16bf a, b0, b1, b2, b3;
        #pragma unroll
        for (int e = 0; e < 8; ++e) {
            a[e]  = alo[e]; a[e + 8]  = ahi[e];
            b0[e] = b0l[e]; b0[e + 8] = b0h[e];
            b1[e] = b1l[e]; b1[e + 8] = b1h[e];
            b2[e] = b2l[e]; b2[e + 8] = b2h[e];
            b3[e] = b3l[e]; b3[e + 8] = b3h[e];
        }

        acc0 = __builtin_amdgcn_wmma_f32_16x16x32_bf16(false, a, false, b0, (short)0, acc0, false, false);
        acc1 = __builtin_amdgcn_wmma_f32_16x16x32_bf16(false, a, false, b1, (short)0, acc1, false, false);
        acc2 = __builtin_amdgcn_wmma_f32_16x16x32_bf16(false, a, false, b2, (short)0, acc2, false, false);
        acc3 = __builtin_amdgcn_wmma_f32_16x16x32_bf16(false, a, false, b3, (short)0, acc3, false, false);
    }

    float* o = out + (size_t)row0 * kNO + col0 + lidx;
    #pragma unroll
    for (int v = 0; v < 8; ++v) {
        int m = v + half * 8;
        o[(size_t)m * kNO +  0] = acc0[v];
        o[(size_t)m * kNO + 16] = acc1[v];
        o[(size_t)m * kNO + 32] = acc2[v];
        o[(size_t)m * kNO + 48] = acc3[v];
    }
}

extern "C" void kernel_launch(void* const* d_in, const int* in_sizes, int n_in,
                              void* d_out, int out_size, void* d_ws, size_t ws_size,
                              hipStream_t stream) {
    const float* x       = (const float*)d_in[0];
    const float* edge    = (const float*)d_in[1];
    const float* Wlin    = (const float*)d_in[2];
    const float* Wout    = (const float*)d_in[3];
    const float* weights = (const float*)d_in[4];
    float* out = (float*)d_out;

    char* ws = (char*)d_ws;
    float*  s    = (float*)ws;   ws += (size_t)kB * kC * sizeof(float);
    float*  ci   = (float*)ws;   ws += (size_t)kB * kC * sizeof(float);
    float*  ssig = (float*)ws;   ws += (size_t)kB * (kH * kH) * sizeof(float);
    __bf16* mid  = (__bf16*)ws;  ws += (size_t)kM * kC * sizeof(__bf16);
    __bf16* Wb   = (__bf16*)ws;

    wcvt_kernel<<<(kNO * kC + 255) / 256, 256, 0, stream>>>(Wout, Wb, kNO * kC);
    agg_kernel<<<kB, kC, 0, stream>>>(x, edge, s);
    ci_kernel<<<kB, kC, 0, stream>>>(s, Wlin, ci);
    score_kernel<<<(kB * kH * kH + 7) / 8, 256, 0, stream>>>(x, ci, ssig);
    pool_kernel<<<kB * kTOK, kC, 0, stream>>>(x, ssig, weights, mid);

    int waves = (kM / 16) * (kNO / 64);   // 1576 * 12 = 18912
    gemm_kernel<<<waves / 8, 256, 0, stream>>>(mid, Wb, out);
}
